// Attention_53034256171713
// MI455X (gfx1250) — matrix operand order repaired and checked by reading, not compiled, not run on hardware
//
#include <hip/hip_runtime.h>
#include <hip/hip_bf16.h>
#include <cstdint>

#define DEVI __device__ __forceinline__

constexpr int   B_    = 8;
constexpr int   N_    = 1024;
constexpr int   C_    = 768;
constexpr int   H_    = 12;
constexpr int   HD_   = 64;
constexpr int   NH_   = B_ * H_;            // 96 (b,h) pairs
constexpr float SCALE_ = 0.125f;            // HD^-0.5
constexpr float LAMBD_ = 4.0f;
constexpr int   ITERS_ = 5;
constexpr long  KV_   = (long)B_ * H_ * N_ * HD_;   // 6291456 (== B*N*C)

typedef __attribute__((ext_vector_type(16))) __bf16 v16bf;
typedef __attribute__((ext_vector_type(8)))  __bf16 v8bf;
typedef __attribute__((ext_vector_type(8)))  float  v8f;
typedef int v4i __attribute__((ext_vector_type(4)));
typedef __attribute__((address_space(1))) v4i v4i_g;   // global
typedef __attribute__((address_space(3))) v4i v4i_l;   // LDS

DEVI __bf16 f2bf(float f) {
  unsigned u = __builtin_bit_cast(unsigned, f);
  u += 0x7FFFu + ((u >> 16) & 1u);                 // round-to-nearest-even
  unsigned short hs = (unsigned short)(u >> 16);
  return __builtin_bit_cast(__bf16, hs);
}

DEVI v16bf cat8(v8bf lo, v8bf hi) {
  return __builtin_shufflevector(lo, hi, 0,1,2,3,4,5,6,7,8,9,10,11,12,13,14,15);
}

// A-fragment (16x32 bf16): lane L holds row m=L&15; half hh=L>>4
// carries K = {k0+16*hh .. +15}
DEVI v16bf load_a(const __bf16* row_ptr, int k0, int hh) {
  v8bf lo = *(const v8bf*)(row_ptr + k0 + 16 * hh);
  v8bf hi = *(const v8bf*)(row_ptr + k0 + 16 * hh + 8);
  return cat8(lo, hi);
}

DEVI v8f wmma_bf16(v16bf a, v16bf b, v8f c) {
  return __builtin_amdgcn_wmma_f32_16x16x32_bf16(false, a, false, b, (short)0, c,
                                                 false, false);
}

// ---------------- CDNA5 async-to-LDS / TDM helpers (probe-guarded) ---------
#if defined(__has_builtin)
#if __has_builtin(__builtin_amdgcn_global_load_async_to_lds_b128)
#define HAS_ASYNC_LDS 1
#endif
#if __has_builtin(__builtin_amdgcn_tensor_load_to_lds)
#define HAS_TDM 1
#endif
#endif

// 16B global -> LDS copy; async (ASYNCcnt) when the builtin exists.
// Builtin prototype: (v4i addrspace(1)*, v4i addrspace(3)*, imm offset, imm cpol)
DEVI void async_cp16(const __bf16* g, __bf16* l) {
#ifdef HAS_ASYNC_LDS
  __builtin_amdgcn_global_load_async_to_lds_b128(
      (v4i_g*)(uintptr_t)g, (v4i_l*)(unsigned)(uintptr_t)l, 0, 0);
#else
  *(v8bf*)l = *(const v8bf*)g;                 // sync fallback
#endif
}

DEVI void wait_async0() {
#ifdef HAS_ASYNC_LDS
#if __has_builtin(__builtin_amdgcn_s_wait_asynccnt)
  __builtin_amdgcn_s_wait_asynccnt(0);
#else
  asm volatile("s_wait_asynccnt 0" ::: "memory");
#endif
#endif
}

#ifdef HAS_TDM
typedef unsigned int u32x4 __attribute__((ext_vector_type(4)));
typedef int          i32x4 __attribute__((ext_vector_type(4)));
typedef int          i32x8 __attribute__((ext_vector_type(8)));

// TDM: load a 2-D tile (tile_dim0=32 keys contiguous, tile_dim1=64 d-rows,
// row stride = N elements) from vT [HD][N] bf16 into LDS [64][32] bf16.
// D# bit layout per CDNA5 ISA ch.8 (group0: flags/lds/global addr, type=2;
// group1: data_size, tensor dims, tile dims, stride).
DEVI void tdm_load_vt_tile(const __bf16* gsrc, const __bf16* ldst) {
  unsigned long long ga = (unsigned long long)(uintptr_t)gsrc;
  unsigned lds = (unsigned)(uintptr_t)ldst;    // low 32 bits = LDS byte offset
  u32x4 g0 = { 1u,                              // count=1, user descriptor
               lds,                             // bits 63:32  lds_addr
               (unsigned)ga,                    // bits 95:64  global_addr lo
               (unsigned)((ga >> 32) & 0x1FFFFFFu) | (2u << 30) }; // addr hi | type=2
  i32x8 g1 = { (int)(1u << 16),                 // data_size = 1 -> 2 bytes
               (int)((unsigned)N_ << 16),       // bits 63:48  tensor_dim0 = 1024
               (int)((unsigned)HD_ << 16),      // bits 95:80  tensor_dim1 = 64
               (int)(32u << 16),                // bits 127:112 tile_dim0 = 32
               (int)HD_,                        // bits 143:128 tile_dim1 = 64
               (int)N_,                         // bits 191:160 tensor_dim0_stride
               0, 0 };
  i32x4 gz = { 0, 0, 0, 0 };
#if __clang_major__ >= 23
  i32x8 gz8 = { 0, 0, 0, 0, 0, 0, 0, 0 };
  __builtin_amdgcn_tensor_load_to_lds(g0, g1, gz, gz, gz8, 0);
#else
  __builtin_amdgcn_tensor_load_to_lds(g0, g1, gz, gz, 0);
#endif
}

DEVI void wait_tensor0() {
#if __has_builtin(__builtin_amdgcn_s_wait_tensorcnt)
  __builtin_amdgcn_s_wait_tensorcnt(0);
#else
  asm volatile("s_wait_tensorcnt 0" ::: "memory");
#endif
}
#endif  // HAS_TDM

// Stage one 32-key step: k2 tile (32x64 bf16, contiguous 4KB) split across
// the 4 waves via async b128; vT tile (64x32 bf16, strided) via TDM from
// wave 0 (fallback: async b128 gather).
DEVI void stage_tiles(const __bf16* k2t, const __bf16* vtt,
                      __bf16* k2d, __bf16* vtd, int wave, int lane) {
#pragma unroll
  for (int t = 0; t < 2; ++t) {
    int idx = (t * 4 + wave) * 32 + lane;        // 0..255 sixteen-byte chunks
    async_cp16(k2t + idx * 8, k2d + idx * 8);
  }
#ifdef HAS_TDM
  if (wave == 0) tdm_load_vt_tile(vtt, vtd);
#else
#pragma unroll
  for (int t = 0; t < 2; ++t) {
    int c = (t * 4 + wave) * 32 + lane;
    int d = c >> 2, sub = c & 3;
    async_cp16(vtt + (long)d * N_ + sub * 8, vtd + c * 8);
  }
#endif
}

// ---------------------------------------------------------------- utilities
__global__ void zero_kernel(float* __restrict__ p, long n) {
  long i = (long)blockIdx.x * blockDim.x + threadIdx.x;
  if (i < n) p[i] = 0.f;
}

__global__ void convert_kernel(const float* __restrict__ in, __bf16* __restrict__ out, long n) {
  long i = (long)blockIdx.x * blockDim.x + threadIdx.x;
  if (i < n) out[i] = f2bf(in[i]);
}

// in [K][Ncols] f32 row-major -> out [Ncols][K] bf16 (B-fragments load contiguously)
__global__ void transpose_convert_kernel(const float* __restrict__ in, __bf16* __restrict__ out,
                                         int K, int Ncols) {
  long i = (long)blockIdx.x * blockDim.x + threadIdx.x;
  if (i >= (long)K * Ncols) return;
  int kk = (int)(i / Ncols), cc = (int)(i % Ncols);
  out[(long)cc * K + kk] = f2bf(in[i]);
}

// ------------------------------------------------------------- QKV GEMM
__global__ __launch_bounds__(256)
void qkv_kernel(const __bf16* __restrict__ xb, const __bf16* __restrict__ wqt,
                const float* __restrict__ b_qkv,
                float* __restrict__ kbuf, __bf16* __restrict__ vtb) {
  const int wave = threadIdx.x >> 5, lane = threadIdx.x & 31;
  const int m = lane & 15, hh = lane >> 4;
  const int TN = (2 * C_) / 16;                  // 96 col tiles
  int tile = blockIdx.x * 8 + wave;
  int i0 = (tile / TN) * 16;
  int j0 = (tile % TN) * 16;
  const __bf16* arow = xb + (long)(i0 + m) * C_;
  const __bf16* bcol = wqt + (long)(j0 + m) * C_;
  v8f acc = {};
  for (int k0 = 0; k0 < C_; k0 += 32) {
    v16bf a = load_a(arow, k0, hh);
    v16bf b = *(const v16bf*)(bcol + k0 + 16 * hh);
    acc = wmma_bf16(a, b, acc);
  }
  int cc = j0 + m;
  float bias = b_qkv[cc];
#pragma unroll
  for (int r = 0; r < 8; ++r) {
    int g = i0 + r + 8 * hh;                     // output row
    int bb = g >> 10, n = g & (N_ - 1);
    float val = acc[r] + bias;
    if (cc < C_) {                               // k
      int h = cc >> 6, d = cc & 63;
      kbuf[(((long)bb * H_ + h) * N_ + n) * HD_ + d] = val;
    } else {                                     // v (store transposed bf16)
      int c2 = cc - C_;
      int h = c2 >> 6, d = c2 & 63;
      vtb[(((long)bb * H_ + h) * HD_ + d) * N_ + n] = f2bf(val);
    }
  }
}

// ------------------------------------------------------------- mu reduction
__global__ __launch_bounds__(256)
void mu_kernel(const float* __restrict__ kbuf, float* __restrict__ mu) {
  __shared__ float red[256];
  int bh = blockIdx.x;
  const float* p = kbuf + (long)bh * N_ * HD_;
  float s = 0.f;
  for (int i = threadIdx.x; i < N_ * HD_; i += 256) s += fabsf(p[i]);
  red[threadIdx.x] = s;
  __syncthreads();
  for (int off = 128; off > 0; off >>= 1) {
    if ((int)threadIdx.x < off) red[threadIdx.x] += red[threadIdx.x + off];
    __syncthreads();
  }
  if (threadIdx.x == 0) mu[bh] = ((float)N_ * C_ / 4.0f) / red[0];
}

// ------------------------------------------------------------- elementwise
__global__ void prep_kernel(const float* __restrict__ k, const float* __restrict__ l,
                            const float* __restrict__ y, const float* __restrict__ mu,
                            float* __restrict__ s, __bf16* __restrict__ k2b) {
  long i = (long)blockIdx.x * blockDim.x + threadIdx.x;
  if (i >= KV_) return;
  int bh = (int)(i >> 16);                       // N*HD == 65536
  float mv = mu[bh];
  float yi = y[i] * (1.0f / mv);
  float a = k[i] - l[i] + yi;
  float t = LAMBD_ * mv;
  float sv = (a >= t) ? (a - t) : ((a <= -t) ? (a + t) : 0.f);
  s[i] = sv;
  k2b[i] = f2bf(k[i] - sv - yi);
}

__global__ void yupd_kernel(const float* __restrict__ k, const float* __restrict__ l,
                            const float* __restrict__ s, const float* __restrict__ mu,
                            float* __restrict__ y) {
  long i = (long)blockIdx.x * blockDim.x + threadIdx.x;
  if (i >= KV_) return;
  int bh = (int)(i >> 16);
  y[i] += mu[bh] * (k[i] - l[i] - s[i]);
}

// ------------------------------------------------------------- attention
// Scores for 32 keys from GLOBAL k2 (used by FINAL max/sum pre-pass).
DEVI void score32_g(const __bf16* k2bh, int jj, int m, int hh,
                    v16bf qa0, v16bf qa1, v8f& st0, v8f& st1) {
  v16bf kb;
  kb  = *(const v16bf*)(k2bh + (long)(jj + m) * HD_ + 16 * hh);
  st0 = wmma_bf16(qa0, kb, st0);
  kb  = *(const v16bf*)(k2bh + (long)(jj + m) * HD_ + 32 + 16 * hh);
  st0 = wmma_bf16(qa1, kb, st0);
  kb  = *(const v16bf*)(k2bh + (long)(jj + 16 + m) * HD_ + 16 * hh);
  st1 = wmma_bf16(qa0, kb, st1);
  kb  = *(const v16bf*)(k2bh + (long)(jj + 16 + m) * HD_ + 32 + 16 * hh);
  st1 = wmma_bf16(qa1, kb, st1);
}

// Scores for 32 keys from the staged LDS tile [32][64].
DEVI void score32_l(const __bf16* k2s, int m, int hh,
                    v16bf qa0, v16bf qa1, v8f& st0, v8f& st1) {
  v16bf kb;
  kb  = *(const v16bf*)(k2s + m * 64 + 16 * hh);
  st0 = wmma_bf16(qa0, kb, st0);
  kb  = *(const v16bf*)(k2s + m * 64 + 32 + 16 * hh);
  st0 = wmma_bf16(qa1, kb, st0);
  kb  = *(const v16bf*)(k2s + (16 + m) * 64 + 16 * hh);
  st1 = wmma_bf16(qa0, kb, st1);
  kb  = *(const v16bf*)(k2s + (16 + m) * 64 + 32 + 16 * hh);
  st1 = wmma_bf16(qa1, kb, st1);
}

// 4 waves/WG, each wave owns a 16-row query tile; online softmax (flash).
// K/V tiles double-buffered in LDS via async-to-LDS + TDM.
// FINAL: exact two-pass softmax + materialize normalized attn to d_out.
template <bool FINAL>
__global__ __launch_bounds__(128)
void attn_kernel(const __bf16* __restrict__ k2b, const __bf16* __restrict__ vt,
                 float* __restrict__ l_out, float* __restrict__ attn_out) {
  __shared__ alignas(16) __bf16 k2s[2][32 * 64];   // 2 x 4KB key tiles
  __shared__ alignas(16) __bf16 vts[2][64 * 32];   // 2 x 4KB v^T tiles
  __shared__ alignas(32) __bf16 ptile[4][16][32];  // per-wave P relayout
  const int wave = threadIdx.x >> 5, lane = threadIdx.x & 31;
  const int m = lane & 15, hh = lane >> 4;
  const int bh = blockIdx.y;
  const int i0 = blockIdx.x * 64 + wave * 16;
  const __bf16* k2bh = k2b + (long)bh * N_ * HD_;
  const __bf16* vtbh = vt + (long)bh * HD_ * N_;

  // query A-fragments: rows i0..i0+15, HD=64 covered by two K=32 fragments
  const __bf16* qrow = k2bh + (long)(i0 + m) * HD_;
  v16bf qa0 = load_a(qrow, 0, hh);
  v16bf qa1 = load_a(qrow, 32, hh);

  float mrow[8], srow[8], invs[8];
#pragma unroll
  for (int r = 0; r < 8; ++r) { mrow[r] = -1e30f; srow[r] = 0.f; }

  if constexpr (FINAL) {
    // pass 1: exact row max + sum over all 1024 keys (cache-hot global reads)
    for (int jj = 0; jj < N_; jj += 32) {
      if (jj + 32 < N_)
        __builtin_prefetch(k2bh + (long)(jj + 32 + m) * HD_, 0, 3);
      v8f st0 = {}, st1 = {};
      score32_g(k2bh, jj, m, hh, qa0, qa1, st0, st1);
#pragma unroll
      for (int r = 0; r < 8; ++r) {
        float s0 = st0[r] * SCALE_, s1 = st1[r] * SCALE_;
        float t = fmaxf(s0, s1);
        t = fmaxf(t, __shfl_xor(t, 1));  t = fmaxf(t, __shfl_xor(t, 2));
        t = fmaxf(t, __shfl_xor(t, 4));  t = fmaxf(t, __shfl_xor(t, 8));
        float mn = fmaxf(mrow[r], t);
        float corr = __expf(mrow[r] - mn);
        float p = __expf(s0 - mn) + __expf(s1 - mn);
        p += __shfl_xor(p, 1);  p += __shfl_xor(p, 2);
        p += __shfl_xor(p, 4);  p += __shfl_xor(p, 8);
        srow[r] = srow[r] * corr + p;
        mrow[r] = mn;
      }
    }
#pragma unroll
    for (int r = 0; r < 8; ++r) invs[r] = 1.f / srow[r];
  }

  v8f acc[4] = {{}, {}, {}, {}};

  stage_tiles(k2bh, vtbh, k2s[0], vts[0], wave, lane);
  const int nst = N_ / 32;
  for (int st = 0; st < nst; ++st) {
    const int jj = st * 32, cur = st & 1;
    wait_async0();
#ifdef HAS_TDM
    if (wave == 0) wait_tensor0();
#endif
    __syncthreads();                             // tiles[cur] visible to all
    if (st + 1 < nst)
      stage_tiles(k2bh + (long)(jj + 32) * HD_, vtbh + jj + 32,
                  k2s[cur ^ 1], vts[cur ^ 1], wave, lane);

    v8f st0 = {}, st1 = {};
    score32_l(k2s[cur], m, hh, qa0, qa1, st0, st1);

    float p0[8], p1[8];
    if constexpr (FINAL) {
      long obase = (long)bh * N_ * N_;
#pragma unroll
      for (int r = 0; r < 8; ++r) {
        p0[r] = __expf(st0[r] * SCALE_ - mrow[r]);
        p1[r] = __expf(st1[r] * SCALE_ - mrow[r]);
        long row = obase + (long)(i0 + r + 8 * hh) * N_ + jj;
        attn_out[row + m]      = p0[r] * invs[r];
        attn_out[row + 16 + m] = p1[r] * invs[r];
      }
    } else {
#pragma unroll
      for (int r = 0; r < 8; ++r) {
        float s0 = st0[r] * SCALE_, s1 = st1[r] * SCALE_;
        float t = fmaxf(s0, s1);
        t = fmaxf(t, __shfl_xor(t, 1));  t = fmaxf(t, __shfl_xor(t, 2));
        t = fmaxf(t, __shfl_xor(t, 4));  t = fmaxf(t, __shfl_xor(t, 8));
        float mn = fmaxf(mrow[r], t);
        float corr = __expf(mrow[r] - mn);
        p0[r] = __expf(s0 - mn);
        p1[r] = __expf(s1 - mn);
        float ps = p0[r] + p1[r];
        ps += __shfl_xor(ps, 1);  ps += __shfl_xor(ps, 2);
        ps += __shfl_xor(ps, 4);  ps += __shfl_xor(ps, 8);
        srow[r] = srow[r] * corr + ps;
        mrow[r] = mn;
#pragma unroll
        for (int nt = 0; nt < 4; ++nt) acc[nt][r] *= corr;
      }
    }

    // P tile (C-layout) -> LDS -> A-fragment layout for P@V
#pragma unroll
    for (int r = 0; r < 8; ++r) {
      int row = r + 8 * hh;
      ptile[wave][row][m]      = f2bf(p0[r]);
      ptile[wave][row][16 + m] = f2bf(p1[r]);
    }
    v8bf plo = *(const v8bf*)&ptile[wave][m][16 * hh];
    v8bf phi = *(const v8bf*)&ptile[wave][m][16 * hh + 8];
    v16bf pa = cat8(plo, phi);
#pragma unroll
    for (int nt = 0; nt < 4; ++nt) {
      v16bf vb = *(const v16bf*)(vts[cur] + (nt * 16 + m) * 32 + 16 * hh);
      acc[nt] = wmma_bf16(pa, vb, acc[nt]);
    }
  }

  if constexpr (!FINAL) {
#pragma unroll
    for (int r = 0; r < 8; ++r) invs[r] = 1.f / srow[r];
  }
  float* lo = l_out + (long)bh * N_ * HD_;
#pragma unroll
  for (int nt = 0; nt < 4; ++nt)
#pragma unroll
    for (int r = 0; r < 8; ++r)
      lo[(long)(i0 + r + 8 * hh) * HD_ + nt * 16 + m] = acc[nt][r] * invs[r];
}

// ------------------------------------------------------------- projection
__global__ __launch_bounds__(256)
void proj_kernel(const __bf16* __restrict__ lb, const __bf16* __restrict__ wpt,
                 const float* __restrict__ b_proj, float* __restrict__ out) {
  const int wave = threadIdx.x >> 5, lane = threadIdx.x & 31;
  const int m = lane & 15, hh = lane >> 4;
  const int TN = C_ / 16;                        // 48
  int tile = blockIdx.x * 8 + wave;
  int i0 = (tile / TN) * 16;
  int j0 = (tile % TN) * 16;
  int g = i0 + m;
  int bb = g >> 10, n = g & (N_ - 1);
  const __bf16* abase = lb + ((long)bb * H_ * N_ + n) * HD_;  // + h*N*HD + d
  const __bf16* bcol = wpt + (long)(j0 + m) * C_;
  v8f acc = {};
  for (int k0 = 0; k0 < C_; k0 += 32) {
    int h = k0 >> 6;                             // k0..k0+31 stays in one head
    int dlo = (k0 & 63) + 16 * hh;
    int dhi = dlo + 8;
    v8bf alo = *(const v8bf*)(abase + (long)h * N_ * HD_ + dlo);
    v8bf ahi = *(const v8bf*)(abase + (long)h * N_ * HD_ + dhi);
    v16bf a = cat8(alo, ahi);
    v16bf b = *(const v16bf*)(bcol + k0 + 16 * hh);
    acc = wmma_bf16(a, b, acc);
  }
  float bias = b_proj[j0 + m];
#pragma unroll
  for (int r = 0; r < 8; ++r) {
    int gr = i0 + r + 8 * hh;
    out[(long)gr * C_ + j0 + m] = acc[r] + bias;
  }
}

// ------------------------------------------------------------- launch
extern "C" void kernel_launch(void* const* d_in, const int* in_sizes, int n_in,
                              void* d_out, int out_size, void* d_ws, size_t ws_size,
                              hipStream_t stream) {
  (void)in_sizes; (void)n_in; (void)out_size; (void)ws_size;
  const float* x      = (const float*)d_in[0];
  const float* W_qkv  = (const float*)d_in[1];
  const float* b_qkv  = (const float*)d_in[2];
  const float* W_proj = (const float*)d_in[3];
  const float* b_proj = (const float*)d_in[4];
  float* out      = (float*)d_out;                   // [B,N,C]
  float* attn_out = out + (size_t)B_ * N_ * C_;      // [B,H,N,N]

  char* w = (char*)d_ws;
  auto carve = [&](size_t bytes) {
    void* p = (void*)w;
    w += (bytes + 255) & ~(size_t)255;
    return p;
  };
  float*  kbuf  = (float*)carve(KV_ * 4);
  float*  lbuf  = (float*)carve(KV_ * 4);
  float*  ybuf  = (float*)carve(KV_ * 4);
  float*  sbuf  = (float*)carve(KV_ * 4);
  float*  mubuf = (float*)carve(NH_ * 4);
  __bf16* xb    = (__bf16*)carve(KV_ * 2);           // B*N*C == KV_
  __bf16* wqt   = (__bf16*)carve((size_t)2 * C_ * C_ * 2);
  __bf16* wpt   = (__bf16*)carve((size_t)C_ * C_ * 2);
  __bf16* vtb   = (__bf16*)carve(KV_ * 2);
  __bf16* k2b   = (__bf16*)carve(KV_ * 2);
  __bf16* lb    = (__bf16*)carve(KV_ * 2);

  const int T = 256;
  const int EB = (int)((KV_ + T - 1) / T);           // elementwise blocks

  convert_kernel<<<EB, T, 0, stream>>>(x, xb, KV_);
  transpose_convert_kernel<<<(int)(((long)C_ * 2 * C_ + T - 1) / T), T, 0, stream>>>(
      W_qkv, wqt, C_, 2 * C_);
  transpose_convert_kernel<<<(int)(((long)C_ * C_ + T - 1) / T), T, 0, stream>>>(
      W_proj, wpt, C_, C_);
  zero_kernel<<<EB, T, 0, stream>>>(lbuf, KV_);
  zero_kernel<<<EB, T, 0, stream>>>(ybuf, KV_);

  qkv_kernel<<<6144, 256, 0, stream>>>(xb, wqt, b_qkv, kbuf, vtb);
  mu_kernel<<<NH_, 256, 0, stream>>>(kbuf, mubuf);

  dim3 ag(N_ / 64, NH_);
  for (int it = 0; it < ITERS_; ++it) {
    prep_kernel<<<EB, T, 0, stream>>>(kbuf, lbuf, ybuf, mubuf, sbuf, k2b);
    attn_kernel<false><<<ag, 128, 0, stream>>>(k2b, vtb, lbuf, nullptr);
    yupd_kernel<<<EB, T, 0, stream>>>(kbuf, lbuf, sbuf, mubuf, ybuf);
  }
  prep_kernel<<<EB, T, 0, stream>>>(kbuf, lbuf, ybuf, mubuf, sbuf, k2b);
  attn_kernel<true><<<ag, 128, 0, stream>>>(k2b, vtb, lbuf, attn_out);

  convert_kernel<<<EB, T, 0, stream>>>(lbuf, lb, KV_);
  proj_kernel<<<3072, 256, 0, stream>>>(lb, wpt, b_proj, out);
}
